// SelfAttention_49718541419145
// MI455X (gfx1250) — compile-verified, matrix operand order checked by reading, not run on hardware
//
#include <hip/hip_runtime.h>
#include <hip/hip_bf16.h>

typedef _Float16 f16x16 __attribute__((ext_vector_type(16)));
typedef _Float16 f16x8  __attribute__((ext_vector_type(8)));
typedef float    f32x8  __attribute__((ext_vector_type(8)));

#define B_   32
#define C_   256
#define N_   1024   // H*W

__device__ __forceinline__ f32x8 wmma16(f16x16 a, f16x16 b, f32x8 c) {
    // D = A(16x32 f16) x B(32x16 f16) + C(16x16 f32)
    return __builtin_amdgcn_wmma_f32_16x16x32_f16(false, a, false, b,
                                                  (short)0, c, false, false);
}

// gfx1250 async global->LDS copy (ASYNCcnt-tracked, no VGPR data path).
// VDST VGPR holds the wave-relative LDS byte offset = low 32 bits of the
// flat shared-memory address (aperture mapping: addr[31:0] == LDS offset).
__device__ __forceinline__ void async_copy_b128(void* lds, const void* gaddr) {
    unsigned l = (unsigned)(unsigned long long)(uintptr_t)lds;
    asm volatile("global_load_async_to_lds_b128 %0, %1, off"
                 : : "v"(l), "v"(gaddr) : "memory");
}
__device__ __forceinline__ void wait_asynccnt0() {
    asm volatile("s_wait_asynccnt 0" ::: "memory");
}

// Fragment loader for both A and B operands.
// A: row-major [16 x K] tile, lane l holds row l%16, K-halfword (l/16)*16..+15.
// B: source is B^T row-major [Nout x K]; lane l holds column l%16 of B over the
//    same K range — identical addressing.
__device__ __forceinline__ f16x16 load_frag(const _Float16* p, int ld) {
    const int lane = threadIdx.x & 31;
    const _Float16* q = p + (lane & 15) * ld + ((lane >> 4) << 4);
    f16x16 f;
    *(f16x8*)&f       = *(const f16x8*)(q);
    *((f16x8*)&f + 1) = *(const f16x8*)(q + 8);
    return f;
}

// ---------------------------------------------------------------- bn stats
__global__ void bn_stats_kernel(const float* __restrict__ x,
                                float* __restrict__ stats) {
    const int c   = blockIdx.x;          // channel
    const int tid = threadIdx.x;         // 256 threads
    float s = 0.f, ss = 0.f;
    for (int b = 0; b < B_; ++b) {
        const float* p = x + ((size_t)b * C_ + c) * N_;
        for (int i = tid; i < N_; i += 256) {
            float v = p[i];
            s += v; ss += v * v;
        }
    }
    __shared__ float ls[256], lss[256];
    ls[tid] = s; lss[tid] = ss;
    __syncthreads();
    for (int off = 128; off > 0; off >>= 1) {
        if (tid < off) { ls[tid] += ls[tid + off]; lss[tid] += lss[tid + off]; }
        __syncthreads();
    }
    if (tid == 0) {
        const float inv_n = 1.0f / (float)(B_ * N_);
        float mean = ls[0] * inv_n;
        float var  = lss[0] * inv_n - mean * mean;
        stats[c]      = mean;
        stats[C_ + c] = rsqrtf(var + 1e-5f);
    }
}

// ---------------------------------------------------------------- weights
__global__ void pack_weights_kernel(const float* __restrict__ Wk,
                                    const float* __restrict__ Wp,
                                    _Float16* __restrict__ wk16,
                                    _Float16* __restrict__ wp16) {
    int i = blockIdx.x * 256 + threadIdx.x;
    if (i < 3 * C_ * C_) wk16[i] = (_Float16)Wk[i];
    if (i < C_ * C_)     wp16[i] = (_Float16)Wp[i];
}

// ------------------------------------------- normalize + transpose to [B,N,C]
__global__ void bn_apply_pack_kernel(const float* __restrict__ x,
                                     const float* __restrict__ stats,
                                     const float* __restrict__ gamma,
                                     const float* __restrict__ beta,
                                     _Float16* __restrict__ t) {
    __shared__ float tile[32][33];
    const int b  = blockIdx.z;
    const int n0 = blockIdx.y * 32;
    const int c0 = blockIdx.x * 32;
    const int tx = threadIdx.x & 31;
    const int ty = threadIdx.x >> 5;          // 0..7
#pragma unroll
    for (int r = 0; r < 4; ++r) {
        int cl = ty + r * 8;
        int c  = c0 + cl;
        float g  = gamma[c] * stats[C_ + c];
        float bt = beta[c] - stats[c] * g;
        float v  = x[(((size_t)b * C_ + c) * N_) + n0 + tx];
        tile[cl][tx] = v * g + bt;
    }
    __syncthreads();
#pragma unroll
    for (int r = 0; r < 4; ++r) {
        int nl = ty + r * 8;
        t[((size_t)b * N_ + n0 + nl) * C_ + c0 + tx] = (_Float16)tile[tx][nl];
    }
}

// ---------------------------------------------------------------- KQV GEMM
// out[M=B*N, 768] = t[M,256] @ Wkqv^T ; q scaled by 1/16, v stored transposed.
// Double-buffered LDS staging via async global->LDS copies.
__global__ void gemm_kqv_kernel(const _Float16* __restrict__ t,
                                const _Float16* __restrict__ w,
                                const float* __restrict__ bias,
                                _Float16* __restrict__ qb,
                                _Float16* __restrict__ kb,
                                _Float16* __restrict__ vT) {
    __shared__ _Float16 sa[2][128 * 40];
    __shared__ _Float16 sb[2][128 * 40];
    const int tid  = threadIdx.x;
    const int lane = tid & 31;
    const int wave = tid >> 5;
    const int wm = wave & 3, wn = wave >> 2;       // 4 x 2 wave grid
    const int n0 = blockIdx.x * 128;
    const int m0 = blockIdx.y * 128;

    auto stage = [&](int k0, int bufi) {
#pragma unroll
        for (int i = 0; i < 2; ++i) {
            int e = (i * 256 + tid) * 8;
            int r = e >> 5, c = e & 31;
            async_copy_b128(&sa[bufi][r * 40 + c],
                            t + (size_t)(m0 + r) * C_ + k0 + c);
            async_copy_b128(&sb[bufi][r * 40 + c],
                            w + (size_t)(n0 + r) * C_ + k0 + c);
        }
    };

    f32x8 acc[2][4];
#pragma unroll
    for (int i = 0; i < 2; ++i)
#pragma unroll
        for (int j = 0; j < 4; ++j)
            acc[i][j] = (f32x8){0.f,0.f,0.f,0.f,0.f,0.f,0.f,0.f};

    stage(0, 0);
    wait_asynccnt0();
    __syncthreads();

    int cur = 0;
    for (int k0 = 0; k0 < C_; k0 += 32) {
        if (k0 + 32 < C_) stage(k0 + 32, cur ^ 1);   // prefetch next tile
        f16x16 af[2], bf[4];
#pragma unroll
        for (int mi = 0; mi < 2; ++mi)
            af[mi] = load_frag(&sa[cur][(wm * 32 + mi * 16) * 40], 40);
#pragma unroll
        for (int ni = 0; ni < 4; ++ni)
            bf[ni] = load_frag(&sb[cur][(wn * 64 + ni * 16) * 40], 40);
#pragma unroll
        for (int mi = 0; mi < 2; ++mi)
#pragma unroll
            for (int ni = 0; ni < 4; ++ni)
                acc[mi][ni] = wmma16(af[mi], bf[ni], acc[mi][ni]);
        wait_asynccnt0();
        __syncthreads();
        cur ^= 1;
    }

#pragma unroll
    for (int mi = 0; mi < 2; ++mi)
#pragma unroll
        for (int ni = 0; ni < 4; ++ni) {
            int j    = n0 + wn * 64 + ni * 16 + (lane & 15);
            int mrow = m0 + wm * 32 + mi * 16 + ((lane >> 4) << 3);
            float bs = bias[j];
            if (j < C_) {                                     // q (pre-scaled)
#pragma unroll
                for (int i = 0; i < 8; ++i)
                    qb[(size_t)(mrow + i) * C_ + j] =
                        (_Float16)((acc[mi][ni][i] + bs) * 0.0625f);
            } else if (j < 2 * C_) {                          // k
#pragma unroll
                for (int i = 0; i < 8; ++i)
                    kb[(size_t)(mrow + i) * C_ + (j - C_)] =
                        (_Float16)(acc[mi][ni][i] + bs);
            } else {                                          // v -> vT[B,C,N]
                int bb = mrow >> 10, nn = mrow & (N_ - 1);
#pragma unroll
                for (int i = 0; i < 8; ++i)
                    vT[((size_t)bb * C_ + (j - 2 * C_)) * N_ + nn + i] =
                        (_Float16)(acc[mi][ni][i] + bs);
            }
        }
}

// ---------------------------------------------------------------- attention
// One block = one batch x 16 query rows. S/P kept in LDS, exact softmax.
__global__ void attention_kernel(const _Float16* __restrict__ qb,
                                 const _Float16* __restrict__ kb,
                                 const _Float16* __restrict__ vT,
                                 _Float16* __restrict__ ob) {
    __shared__ _Float16 sq[16 * 264];    // q rows   (ld 264)
    __shared__ _Float16 sp[16 * 1032];   // S then P (ld 1032)
    __shared__ float red[256];
    __shared__ float rowmax[16], rowinv[16];

    const int tid  = threadIdx.x;
    const int lane = tid & 31;
    const int wave = tid >> 5;
    const int b  = blockIdx.y;
    const int r0 = blockIdx.x * 16;

    {   // stage the 16 q rows into LDS (async copies, no VGPR bounce)
        int e = tid * 16;
        int r = e >> 8, c = e & 255;
        const _Float16* src = qb + ((size_t)b * N_ + r0 + r) * C_ + c;
        async_copy_b128(&sq[r * 264 + c], src);
        async_copy_b128(&sq[r * 264 + c + 8], src + 8);
    }
    wait_asynccnt0();
    __syncthreads();

    // ---- pass 1: S = q k^T  (each wave owns 8 column tiles of 16)
    f16x16 af[8];
#pragma unroll
    for (int kk = 0; kk < 8; ++kk) af[kk] = load_frag(&sq[kk * 32], 264);
    for (int mt = wave * 8; mt < wave * 8 + 8; ++mt) {
        f32x8 s = (f32x8){0.f,0.f,0.f,0.f,0.f,0.f,0.f,0.f};
        const _Float16* kbase = kb + ((size_t)b * N_ + mt * 16) * C_;
#pragma unroll
        for (int kk = 0; kk < 8; ++kk)
            s = wmma16(af[kk], load_frag(kbase + kk * 32, C_), s);
        int col = mt * 16 + (lane & 15);
        int rhi = (lane >> 4) << 3;
#pragma unroll
        for (int i = 0; i < 8; ++i)
            sp[(i + rhi) * 1032 + col] = (_Float16)s[i];
    }
    __syncthreads();

    // ---- row max (16 threads per row, 64 elements each)
    {
        int row = tid >> 4, ch = tid & 15;
        const _Float16* p = &sp[row * 1032 + ch * 64];
        float m = -3.0e38f;
        for (int e = 0; e < 64; ++e) m = fmaxf(m, (float)p[e]);
        red[row * 16 + ch] = m;
    }
    __syncthreads();
    if (tid < 16) {
        float m = -3.0e38f;
        for (int j = 0; j < 16; ++j) m = fmaxf(m, red[tid * 16 + j]);
        rowmax[tid] = m;
    }
    __syncthreads();

    // ---- exp in place + row sum
    {
        int row = tid >> 4, ch = tid & 15;
        _Float16* p = &sp[row * 1032 + ch * 64];
        float mx = rowmax[row], sum = 0.f;
        for (int e = 0; e < 64; ++e) {
            float v = __expf((float)p[e] - mx);
            p[e] = (_Float16)v;
            sum += v;
        }
        red[row * 16 + ch] = sum;
    }
    __syncthreads();
    if (tid < 16) {
        float sum = 0.f;
        for (int j = 0; j < 16; ++j) sum += red[tid * 16 + j];
        rowinv[tid] = 1.0f / sum;
    }
    __syncthreads();

    // ---- pass 2: O = P V   (wave owns 32 output channels)
    const int c0 = wave * 32;
    f32x8 o0 = (f32x8){0.f,0.f,0.f,0.f,0.f,0.f,0.f,0.f};
    f32x8 o1 = o0;
    const _Float16* v0 = vT + ((size_t)b * C_ + c0) * N_;
    const _Float16* v1 = vT + ((size_t)b * C_ + c0 + 16) * N_;
    for (int ms = 0; ms < N_; ms += 32) {
        f16x16 ap  = load_frag(&sp[ms], 1032);
        o0 = wmma16(ap, load_frag(v0 + ms, N_), o0);
        o1 = wmma16(ap, load_frag(v1 + ms, N_), o1);
    }
    {
        int rhi = (lane >> 4) << 3;
        int cA = c0 + (lane & 15), cB = cA + 16;
#pragma unroll
        for (int i = 0; i < 8; ++i) {
            float inv = rowinv[i + rhi];
            size_t rowi = (size_t)b * N_ + r0 + i + rhi;
            ob[rowi * C_ + cA] = (_Float16)(o0[i] * inv);
            ob[rowi * C_ + cB] = (_Float16)(o1[i] * inv);
        }
    }
}

// ---------------------------------------------------------------- proj GEMM
// out[B,C,H,W] = o @ Wproj^T + bias + x   (fp32 epilogue, residual fused)
__global__ void gemm_proj_kernel(const _Float16* __restrict__ o,
                                 const _Float16* __restrict__ w,
                                 const float* __restrict__ bias,
                                 const float* __restrict__ x,
                                 float* __restrict__ out) {
    __shared__ _Float16 sa[2][128 * 40];
    __shared__ _Float16 sb[2][128 * 40];
    const int tid  = threadIdx.x;
    const int lane = tid & 31;
    const int wave = tid >> 5;
    const int wm = wave & 3, wn = wave >> 2;
    const int n0 = blockIdx.x * 128;
    const int m0 = blockIdx.y * 128;

    auto stage = [&](int k0, int bufi) {
#pragma unroll
        for (int i = 0; i < 2; ++i) {
            int e = (i * 256 + tid) * 8;
            int r = e >> 5, c = e & 31;
            async_copy_b128(&sa[bufi][r * 40 + c],
                            o + (size_t)(m0 + r) * C_ + k0 + c);
            async_copy_b128(&sb[bufi][r * 40 + c],
                            w + (size_t)(n0 + r) * C_ + k0 + c);
        }
    };

    f32x8 acc[2][4];
#pragma unroll
    for (int i = 0; i < 2; ++i)
#pragma unroll
        for (int j = 0; j < 4; ++j)
            acc[i][j] = (f32x8){0.f,0.f,0.f,0.f,0.f,0.f,0.f,0.f};

    stage(0, 0);
    wait_asynccnt0();
    __syncthreads();

    int cur = 0;
    for (int k0 = 0; k0 < C_; k0 += 32) {
        if (k0 + 32 < C_) stage(k0 + 32, cur ^ 1);
        f16x16 af[2], bf[4];
#pragma unroll
        for (int mi = 0; mi < 2; ++mi)
            af[mi] = load_frag(&sa[cur][(wm * 32 + mi * 16) * 40], 40);
#pragma unroll
        for (int ni = 0; ni < 4; ++ni)
            bf[ni] = load_frag(&sb[cur][(wn * 64 + ni * 16) * 40], 40);
#pragma unroll
        for (int mi = 0; mi < 2; ++mi)
#pragma unroll
            for (int ni = 0; ni < 4; ++ni)
                acc[mi][ni] = wmma16(af[mi], bf[ni], acc[mi][ni]);
        wait_asynccnt0();
        __syncthreads();
        cur ^= 1;
    }

#pragma unroll
    for (int mi = 0; mi < 2; ++mi)
#pragma unroll
        for (int ni = 0; ni < 4; ++ni) {
            int c    = n0 + wn * 64 + ni * 16 + (lane & 15);
            int mrow = m0 + wm * 32 + mi * 16 + ((lane >> 4) << 3);
            int bb = mrow >> 10, nn = mrow & (N_ - 1);
            float bs = bias[c];
            size_t base = ((size_t)bb * C_ + c) * N_ + nn;
#pragma unroll
            for (int i = 0; i < 8; ++i)
                out[base + i] = acc[mi][ni][i] + bs + x[base + i];
        }
}

// ---------------------------------------------------------------- launcher
extern "C" void kernel_launch(void* const* d_in, const int* in_sizes, int n_in,
                              void* d_out, int out_size, void* d_ws, size_t ws_size,
                              hipStream_t stream) {
    (void)in_sizes; (void)n_in; (void)out_size; (void)ws_size;
    const float* x     = (const float*)d_in[0];
    const float* gamma = (const float*)d_in[1];
    const float* beta  = (const float*)d_in[2];
    const float* Wkqv  = (const float*)d_in[3];
    const float* bkqv  = (const float*)d_in[4];
    const float* Wproj = (const float*)d_in[5];
    const float* bproj = (const float*)d_in[6];
    float* out = (float*)d_out;

    const size_t M = (size_t)B_ * N_;              // 32768
    char* ws = (char*)d_ws;
    float*    stats = (float*)ws;      ws += 4096;
    _Float16* wk16  = (_Float16*)ws;   ws += (size_t)3 * C_ * C_ * 2;
    _Float16* wp16  = (_Float16*)ws;   ws += (size_t)C_ * C_ * 2;
    _Float16* tbuf  = (_Float16*)ws;   ws += M * C_ * 2;   // reused as o-buffer
    _Float16* qb    = (_Float16*)ws;   ws += M * C_ * 2;
    _Float16* kb    = (_Float16*)ws;   ws += M * C_ * 2;
    _Float16* vT    = (_Float16*)ws;   ws += M * C_ * 2;

    bn_stats_kernel     <<<C_, 256, 0, stream>>>(x, stats);
    pack_weights_kernel <<<(3 * C_ * C_) / 256, 256, 0, stream>>>(Wkqv, Wproj, wk16, wp16);
    bn_apply_pack_kernel<<<dim3(C_ / 32, N_ / 32, B_), 256, 0, stream>>>(x, stats, gamma, beta, tbuf);
    gemm_kqv_kernel     <<<dim3(6, M / 128), 256, 0, stream>>>(tbuf, wk16, bkqv, qb, kb, vT);
    attention_kernel    <<<dim3(N_ / 16, B_), 256, 0, stream>>>(qb, kb, vT, tbuf);
    gemm_proj_kernel    <<<dim3(2, M / 128), 256, 0, stream>>>(tbuf, wp16, bproj, x, out);
}
